// DeepGCN_slic_V3_23845658427413
// MI455X (gfx1250) — matrix operand order repaired and checked by reading, not compiled, not run on hardware
//
#include <hip/hip_runtime.h>
#include <math.h>

// ---------------- model constants ----------------
#define BATCH 32
#define IMGSZ 64
#define CCH   192
#define NROWS 14
#define NSEG  (NROWS*NROWS)     // 196
#define NBLOCKS 12
#define NCLS  1000
#define KMAX  18
#define BNEPS 1e-5f

typedef __attribute__((ext_vector_type(16))) __bf16 bf16x16;
typedef __attribute__((ext_vector_type(4)))  __bf16 bf16x4;
typedef __attribute__((ext_vector_type(8)))  float  f32x8;
typedef long long ll;

// =====================================================================
// WMMA bf16 GEMM:  Y[z] = A[z] (MxK) * X[z] (KxN) + bias   (all fp32 in
// memory; converted to bf16 while staging into LDS; f32 accumulate).
// Batched over gridDim.z with element strides sA/sX/sY (0 => shared).
// Tile: 128x128 per 128-thread workgroup (4 waves, each owning a
// 64x64 sub-tile = 16 accumulators), K-step 32, double-buffered LDS,
// v_wmma_f32_16x16x32_bf16.  Square per-wave tiles maximize operand
// reuse: 8 fragment loads feed 16 WMMAs (0.5KB LDS read per WMMA).
// Staging is branchless (clamped-address loads + select) and uses
// float4 (global_load_b128) fast paths when the slab is full.
// =====================================================================
#define TM 128
#define TN 128
#define TK 32
#define LDSK 40   // padded row stride (80B, 16B-aligned vectors, no bank conflicts)
#define GTHR 128  // threads per GEMM block (4 waves)

__global__ __launch_bounds__(GTHR, 1)
void gemm_bf16_wmma(const float* __restrict__ A, ll sA,
                    const float* __restrict__ X, ll sX,
                    float* __restrict__ Y, ll sY,
                    const float* __restrict__ bias,
                    int M, int N, int K)
{
    __shared__ __align__(16) __bf16 As[2][TM][LDSK]; // [m][k]
    __shared__ __align__(16) __bf16 Bs[2][TN][LDSK]; // [n][k]  (X staged transposed)

    const int z = blockIdx.z;
    A += (ll)z * sA;  X += (ll)z * sX;  Y += (ll)z * sY;

    const int m0 = blockIdx.y * TM;
    const int n0 = blockIdx.x * TN;
    const int tid  = threadIdx.x;
    const int lane = tid & 31;
    const int wave = tid >> 5;
    const int half = lane >> 4;       // lane 0-15 vs 16-31
    const int lrow = lane & 15;
    const int wm = wave >> 1;         // 2 waves along M  (64-row strips)
    const int wn = wave & 1;          // 2 waves along N  (64-col strips)

    f32x8 acc[4][4];
#pragma unroll
    for (int i = 0; i < 4; ++i)
#pragma unroll
        for (int j = 0; j < 4; ++j)
#pragma unroll
            for (int v = 0; v < 8; ++v) acc[i][j][v] = 0.0f;

    const int nk = (K + TK - 1) / TK;

    auto stage = [&](int buf, int k0) {
        // ---- A tile 128x32 (fp32 -> bf16) ----
        if (((K & 3) == 0) && (k0 + TK <= K)) {
            // vector fast path: full K slab, 16B-aligned rows
#pragma unroll
            for (int i = 0; i < 8; ++i) {
                int e = tid + i * GTHR;           // 0..1023 float4 slots
                int r = e >> 3, c4 = (e & 7) << 2;
                int gm = m0 + r;
                int gmc = gm < M ? gm : (M - 1);  // clamped, always valid
                float4 v = *(const float4*)&A[(ll)gmc * K + k0 + c4];
                if (gm >= M) { v.x = 0.f; v.y = 0.f; v.z = 0.f; v.w = 0.f; }
                bf16x4 h = { (__bf16)v.x, (__bf16)v.y, (__bf16)v.z, (__bf16)v.w };
                *(bf16x4*)&As[buf][r][c4] = h;
            }
        } else {
            // scalar path (K tails / tiny-K layers), branchless clamped loads
#pragma unroll
            for (int i = 0; i < 32; ++i) {
                int e = tid + i * GTHR;           // 0..4095
                int r = e >> 5, c = e & 31;
                int gm = m0 + r, gk = k0 + c;
                int gmc = gm < M ? gm : (M - 1);
                int gkc = gk < K ? gk : (K - 1);
                float v = A[(ll)gmc * K + gkc];
                if (gm >= M || gk >= K) v = 0.f;
                As[buf][r][c] = (__bf16)v;
            }
        }
        // ---- X tile 32x128, stored N-major (transposed) in LDS ----
        if (((N & 3) == 0) && (k0 + TK <= K)) {
#pragma unroll
            for (int i = 0; i < 8; ++i) {
                int e = tid + i * GTHR;           // 0..1023 float4 slots
                int r = e >> 5, c4 = (e & 31) << 2;   // r = k (always < K), c4 = n offset
                int gn = n0 + c4;
                int gnc = gn < N ? gn : (N - 4);  // N%4==0 -> aligned & valid
                float4 v = *(const float4*)&X[(ll)(k0 + r) * N + gnc];
                if (gn >= N) { v.x = 0.f; v.y = 0.f; v.z = 0.f; v.w = 0.f; }
                Bs[buf][c4 + 0][r] = (__bf16)v.x;
                Bs[buf][c4 + 1][r] = (__bf16)v.y;
                Bs[buf][c4 + 2][r] = (__bf16)v.z;
                Bs[buf][c4 + 3][r] = (__bf16)v.w;
            }
        } else {
#pragma unroll
            for (int i = 0; i < 32; ++i) {
                int e = tid + i * GTHR;
                int r = e >> 7, c = e & 127;      // r = k, c = n (coalesced global read)
                int gk = k0 + r, gn = n0 + c;
                int gkc = gk < K ? gk : (K - 1);
                int gnc = gn < N ? gn : (N - 1);
                float v = X[(ll)gkc * N + gnc];
                if (gk >= K || gn >= N) v = 0.f;
                Bs[buf][c][r] = (__bf16)v;
            }
        }
        // speculative prefetch two K-steps ahead -> global_prefetch_b8
        int kp = k0 + 2 * TK;
        if (kp < K) {
            int pr = m0 + tid; if (pr >= M) pr = M - 1;
            __builtin_prefetch(&A[(ll)pr * K + kp], 0, 1);
            __builtin_prefetch(&X[(ll)kp * N + n0 + tid], 0, 1);
        }
    };

    stage(0, 0);
    __syncthreads();

    for (int kt = 0; kt < nk; ++kt) {
        const int buf = kt & 1;
        if (kt + 1 < nk) stage(buf ^ 1, (kt + 1) * TK);

        // --- load fragments (ISA 7.12.2 16-bit layouts) ---
        bf16x16 afrag[4], bfrag[4];
#pragma unroll
        for (int i = 0; i < 4; ++i) {
            int row = wm * 64 + i * 16 + lrow;          // M index
            const __bf16* pr = &As[buf][row][half * 8]; // lanes0-15: K 0..7 / lanes16-31: K 8..15
            bf16x16 f;
#pragma unroll
            for (int q = 0; q < 8; ++q) f[q] = pr[q];
#pragma unroll
            for (int q = 0; q < 8; ++q) f[8 + q] = pr[16 + q]; // +16 K offset
            afrag[i] = f;
        }
#pragma unroll
        for (int j = 0; j < 4; ++j) {
            int col = wn * 64 + j * 16 + lrow;           // N index
            const __bf16* pr = &Bs[buf][col][half * 8];  // lanes0-15: K 0..7,16..23 / 16-31: K 8..15,24..31
            bf16x16 f;
#pragma unroll
            for (int q = 0; q < 8; ++q) f[q] = pr[q];
#pragma unroll
            for (int q = 0; q < 8; ++q) f[8 + q] = pr[16 + q];
            bfrag[j] = f;
        }
#pragma unroll
        for (int i = 0; i < 4; ++i)
#pragma unroll
            for (int j = 0; j < 4; ++j)
                acc[i][j] = __builtin_amdgcn_wmma_f32_16x16x32_bf16(
                    false, afrag[i], false, bfrag[j], (short)0, acc[i][j], false, false);

        __syncthreads();
    }

    // epilogue: D 16x16 layout -> lane0-15: M=v, lane16-31: M=v+8, N=lrow
#pragma unroll
    for (int i = 0; i < 4; ++i)
#pragma unroll
        for (int j = 0; j < 4; ++j) {
            int col = n0 + wn * 64 + j * 16 + lrow;
            if (col >= N) continue;
#pragma unroll
            for (int v = 0; v < 8; ++v) {
                int row = m0 + wm * 64 + i * 16 + half * 8 + v;
                if (row < M) {
                    float o = acc[i][j][v];
                    if (bias) o += bias[row];
                    Y[(ll)row * N + col] = o;
                }
            }
        }
}

// =====================================================================
// BatchNorm: stats over (B, spatial) per channel, then affine apply
// =====================================================================
__global__ void bn_stats_kernel(const float* __restrict__ x, float* __restrict__ mean,
                                float* __restrict__ rinv, int Bb, int Cc, int Nn)
{
    const int c = blockIdx.x;
    __shared__ float s1s[256], s2s[256];
    float s1 = 0.f, s2 = 0.f;
    const int total = Bb * Nn;
    for (int i = threadIdx.x; i < total; i += 256) {
        int b = i / Nn, n = i - b * Nn;
        float v = x[((ll)b * Cc + c) * Nn + n];
        s1 += v; s2 += v * v;
    }
    s1s[threadIdx.x] = s1; s2s[threadIdx.x] = s2;
    __syncthreads();
    for (int off = 128; off; off >>= 1) {
        if (threadIdx.x < off) {
            s1s[threadIdx.x] += s1s[threadIdx.x + off];
            s2s[threadIdx.x] += s2s[threadIdx.x + off];
        }
        __syncthreads();
    }
    if (threadIdx.x == 0) {
        float m = s1s[0] / total;
        float var = s2s[0] / total - m * m;
        mean[c] = m;
        rinv[c] = rsqrtf(var + BNEPS);
    }
}

__global__ void bn_apply_kernel(const float* __restrict__ x, float* __restrict__ y,
                                const float* __restrict__ mean, const float* __restrict__ rinv,
                                const float* __restrict__ g, const float* __restrict__ bb,
                                const float* __restrict__ res, int relu,
                                int Cc, int Nn, ll total)
{
    ll i = (ll)blockIdx.x * 256 + threadIdx.x;
    if (i >= total) return;
    int c = (int)((i / Nn) % Cc);
    float v = (x[i] - mean[c]) * rinv[c] * g[c] + bb[c];
    if (res) v += res[i];
    if (relu) v = fmaxf(v, 0.0f);
    y[i] = v;
}

// =====================================================================
// L2-normalize feature columns; write fn (C,N) and fnT (N,C) per batch
// =====================================================================
__global__ void l2norm_kernel(const float* __restrict__ f, float* __restrict__ fn,
                              float* __restrict__ fnT, int Cc, int Nn, int total /*B*N*/)
{
    int i = blockIdx.x * 256 + threadIdx.x;
    if (i >= total) return;
    int b = i / Nn, n = i - b * Nn;
    const float* base = f + (ll)b * Cc * Nn + n;
    float s = 0.f;
    for (int c = 0; c < Cc; ++c) { float v = base[(ll)c * Nn]; s += v * v; }
    float inv = 1.0f / fmaxf(sqrtf(s), 1e-12f);
    for (int c = 0; c < Cc; ++c) {
        float v = base[(ll)c * Nn] * inv;
        fn[((ll)b * Cc + c) * Nn + n] = v;
        fnT[((ll)b * Nn + n) * Cc + c] = v;
    }
}

// =====================================================================
// top-k indices per sim row (descending, duplicates resolved by index)
// =====================================================================
__global__ void topk_kernel(const float* __restrict__ sim, int* __restrict__ idx,
                            int Nn, int k, int total /*B*N*/)
{
    int i = blockIdx.x * 256 + threadIdx.x;
    if (i >= total) return;
    const float* row = sim + (ll)i * Nn;
    float prevV = INFINITY; int prevI = -1;
    for (int j = 0; j < k; ++j) {
        float best = -INFINITY; int bi = 0;
        for (int m = 0; m < Nn; ++m) {
            float v = row[m];
            bool after = (v < prevV) || (v == prevV && m > prevI);
            if (!after) continue;
            if (v > best) { best = v; bi = m; }
        }
        idx[(ll)i * KMAX + j] = bi;
        prevV = best; prevI = bi;
    }
}

// =====================================================================
// max-relative gather + interleaved concat: z[2c]=f, z[2c+1]=max_j f[idx]-f
// =====================================================================
__global__ void mr_concat_kernel(const float* __restrict__ f, const int* __restrict__ idx,
                                 float* __restrict__ z, int Cc, int Nn, int k, ll total)
{
    ll i = (ll)blockIdx.x * 256 + threadIdx.x;
    if (i >= total) return;
    int n = (int)(i % Nn);
    int c = (int)((i / Nn) % Cc);
    int b = (int)(i / ((ll)Cc * Nn));
    float self = f[i];
    const int* ip = idx + ((ll)b * Nn + n) * KMAX;
    float m = -INFINITY;
    for (int j = 0; j < k; ++j)
        m = fmaxf(m, f[((ll)b * Cc + c) * Nn + ip[j]]);
    m -= self;
    z[((ll)b * 2 * Cc + 2 * c) * Nn + n]     = self;
    z[((ll)b * 2 * Cc + 2 * c + 1) * Nn + n] = m;
}

// =====================================================================
// direct 3x3 conv (stem path, ~4% of FLOPs)
// =====================================================================
__global__ void conv3x3_kernel(const float* __restrict__ in, const float* __restrict__ w,
                               const float* __restrict__ bias, float* __restrict__ out,
                               int Ci, int H, int W, int Co, int Ho, int Wo,
                               int stride, int pad, ll total)
{
    ll i = (ll)blockIdx.x * 256 + threadIdx.x;
    if (i >= total) return;
    int ox = (int)(i % Wo);
    int oy = (int)((i / Wo) % Ho);
    int co = (int)((i / ((ll)Wo * Ho)) % Co);
    int b  = (int)(i / ((ll)Wo * Ho * Co));
    float s = bias[co];
    for (int ci = 0; ci < Ci; ++ci) {
        const float* ip = in + ((ll)b * Ci + ci) * H * W;
        const float* wp = w + ((ll)co * Ci + ci) * 9;
        for (int ky = 0; ky < 3; ++ky) {
            int iy = oy * stride - pad + ky;
            if ((unsigned)iy >= (unsigned)H) continue;
            for (int kx = 0; kx < 3; ++kx) {
                int ix = ox * stride - pad + kx;
                if ((unsigned)ix >= (unsigned)W) continue;
                s += ip[iy * W + ix] * wp[ky * 3 + kx];
            }
        }
    }
    out[i] = s;
}

__global__ void add_pos_kernel(float* __restrict__ x, const float* __restrict__ pe,
                               int CN, ll total)
{
    ll i = (ll)blockIdx.x * 256 + threadIdx.x;
    if (i >= total) return;
    x[i] += pe[i % CN];
}

// =====================================================================
// segment feature extraction (compute_features)
// =====================================================================
__global__ void features_kernel(const float* __restrict__ img, const int* __restrict__ seg,
                                float* __restrict__ out)
{
    const int b = blockIdx.x;
    __shared__ float sCnt[NSEG], sX[NSEG], sY[NSEG], sV[NSEG], sV2[NSEG];
    for (int s = threadIdx.x; s < NSEG; s += 256) {
        sCnt[s] = 0.f; sX[s] = 0.f; sY[s] = 0.f; sV[s] = 0.f; sV2[s] = 0.f;
    }
    __syncthreads();
    const int HW = IMGSZ * IMGSZ;
    for (int p = threadIdx.x; p < HW; p += 256) {
        int y = p / IMGSZ, x = p - y * IMGSZ;
        int s = seg[(ll)b * HW + p];
        float v = img[((ll)b * 3 + 0) * HW + p];
        atomicAdd(&sCnt[s], 1.0f);
        atomicAdd(&sX[s], (float)x);
        atomicAdd(&sY[s], (float)y);
        atomicAdd(&sV[s], v);
        atomicAdd(&sV2[s], v * v);
    }
    __syncthreads();
    for (int s = threadIdx.x; s < NSEG; s += 256) {
        float cnt = sCnt[s];
        bool empty = (cnt <= 0.f);
        float xc = empty ? 0.f : sX[s] / cnt;
        float yc = empty ? 0.f : sY[s] / cnt;
        float m0 = sV[s] / (float)HW;
        float var0 = (sV2[s] - (float)HW * m0 * m0) / (float)(HW - 1);
        float std0 = sqrtf(fmaxf(var0, 1e-12f));
        int xr = (int)rintf(xc), yr = (int)rintf(yc);
        float r = 0.f, gch = 0.f, bl = 0.f;
        if (!empty) {
            r   = img[((ll)b * 3 + 0) * HW + yr * IMGSZ + xr];
            gch = img[((ll)b * 3 + 1) * HW + yr * IMGSZ + xr];
            bl  = img[((ll)b * 3 + 2) * HW + yr * IMGSZ + xr];
        }
        float vals[11] = {xc, yc, m0, m0, m0, std0, std0, std0, r, gch, bl};
        for (int ch = 0; ch < 11; ++ch)
            out[((ll)b * 11 + ch) * NSEG + s] = vals[ch];
    }
}

// =====================================================================
// head: means -> xm (C, B); final transpose into d_out
// =====================================================================
__global__ void head_mean_kernel(const float* __restrict__ x1, const float* __restrict__ x2,
                                 float* __restrict__ xm, int Cc, int Nn, int Bb)
{
    int i = blockIdx.x * 256 + threadIdx.x;
    if (i >= Bb * Cc) return;
    int b = i / Cc, c = i - b * Cc;
    float s1 = 0.f, s2 = 0.f;
    const float* p1 = x1 + ((ll)b * Cc + c) * Nn;
    const float* p2 = x2 + ((ll)b * Cc + c) * Nn;
    for (int n = 0; n < Nn; ++n) { s1 += p1[n]; s2 += p2[n]; }
    xm[(ll)c * Bb + b] = 0.5f * s1 / Nn + 1.5f * s2 / Nn;
}

__global__ void final_out_kernel(const float* __restrict__ p2, float* __restrict__ out)
{
    int i = blockIdx.x * 256 + threadIdx.x;
    if (i >= BATCH * NCLS) return;
    int b = i / NCLS, o = i - b * NCLS;
    out[i] = p2[(ll)o * BATCH + b];
}

// =====================================================================
// host side
// =====================================================================
struct CP  { const float *w, *b, *g, *bb; };
struct Blk { CP fc1, mr, fc2, ffn1, ffn2; };

static CP readCP(void* const* d_in, int& p) {
    CP c;
    c.w  = (const float*)d_in[p++];
    c.b  = (const float*)d_in[p++];
    c.g  = (const float*)d_in[p++];
    c.bb = (const float*)d_in[p++];
    return c;
}

static void launch_gemm(const float* A, ll sA, const float* X, ll sX,
                        float* Y, ll sY, const float* bias,
                        int M, int N, int K, int Z, hipStream_t s)
{
    dim3 g((N + TN - 1) / TN, (M + TM - 1) / TM, Z);
    gemm_bf16_wmma<<<g, GTHR, 0, s>>>(A, sA, X, sX, Y, sY, bias, M, N, K);
}

static void launch_bn(const float* x, float* y, const float* g, const float* bb,
                      const float* res, int relu, int Bb, int Cc, int Nn,
                      float* bnm, float* bnr, hipStream_t s)
{
    bn_stats_kernel<<<Cc, 256, 0, s>>>(x, bnm, bnr, Bb, Cc, Nn);
    ll total = (ll)Bb * Cc * Nn;
    bn_apply_kernel<<<(int)((total + 255) / 256), 256, 0, s>>>(x, y, bnm, bnr, g, bb, res, relu, Cc, Nn, total);
}

struct WS {
    float *featA, *x1, *x2, *y, *fn, *fnT, *sim, *big, *tA, *tB, *xm, *p1, *p2, *bnm, *bnr;
    int* idx;
};

static void run_block(float* x, const Blk& P, int k, WS& ws, hipStream_t s)
{
    const int Nn = NSEG, Cc = CCH, Bb = BATCH;
    // grapher
    launch_gemm(P.fc1.w, 0, x, (ll)Cc * Nn, ws.y, (ll)Cc * Nn, P.fc1.b, Cc, Nn, Cc, Bb, s);
    launch_bn(ws.y, ws.y, P.fc1.g, P.fc1.bb, nullptr, 0, Bb, Cc, Nn, ws.bnm, ws.bnr, s);
    l2norm_kernel<<<(Bb * Nn + 255) / 256, 256, 0, s>>>(ws.y, ws.fn, ws.fnT, Cc, Nn, Bb * Nn);
    launch_gemm(ws.fnT, (ll)Nn * Cc, ws.fn, (ll)Cc * Nn, ws.sim, (ll)Nn * Nn, nullptr, Nn, Nn, Cc, Bb, s);
    topk_kernel<<<(Bb * Nn + 255) / 256, 256, 0, s>>>(ws.sim, ws.idx, Nn, k, Bb * Nn);
    float* z2 = ws.big;
    float* z3 = ws.big + (ll)Bb * 2 * Cc * Nn;
    {
        ll t = (ll)Bb * Cc * Nn;
        mr_concat_kernel<<<(int)((t + 255) / 256), 256, 0, s>>>(ws.y, ws.idx, z2, Cc, Nn, k, t);
    }
    launch_gemm(P.mr.w, 0, z2, (ll)2 * Cc * Nn, z3, (ll)2 * Cc * Nn, P.mr.b, 2 * Cc, Nn, 2 * Cc, Bb, s);
    launch_bn(z3, z3, P.mr.g, P.mr.bb, nullptr, 1, Bb, 2 * Cc, Nn, ws.bnm, ws.bnr, s);
    launch_gemm(P.fc2.w, 0, z3, (ll)2 * Cc * Nn, ws.y, (ll)Cc * Nn, P.fc2.b, Cc, Nn, 2 * Cc, Bb, s);
    launch_bn(ws.y, x, P.fc2.g, P.fc2.bb, x, 0, Bb, Cc, Nn, ws.bnm, ws.bnr, s);
    // ffn
    float* t4 = ws.big;
    launch_gemm(P.ffn1.w, 0, x, (ll)Cc * Nn, t4, (ll)4 * Cc * Nn, P.ffn1.b, 4 * Cc, Nn, Cc, Bb, s);
    launch_bn(t4, t4, P.ffn1.g, P.ffn1.bb, nullptr, 1, Bb, 4 * Cc, Nn, ws.bnm, ws.bnr, s);
    launch_gemm(P.ffn2.w, 0, t4, (ll)4 * Cc * Nn, ws.y, (ll)Cc * Nn, P.ffn2.b, Cc, Nn, 4 * Cc, Bb, s);
    launch_bn(ws.y, x, P.ffn2.g, P.ffn2.bb, x, 0, Bb, Cc, Nn, ws.bnm, ws.bnr, s);
}

extern "C" void kernel_launch(void* const* d_in, const int* in_sizes, int n_in,
                              void* d_out, int out_size, void* d_ws, size_t ws_size,
                              hipStream_t stream)
{
    (void)in_sizes; (void)n_in; (void)out_size; (void)ws_size;
    int p = 0;
    const float* inputs = (const float*)d_in[p++];
    const int*   segmaps = (const int*)d_in[p++];
    CP slic[5]; for (int i = 0; i < 5; ++i) slic[i] = readCP(d_in, p);
    CP tiny[5]; for (int i = 0; i < 5; ++i) tiny[i] = readCP(d_in, p);
    const float* pos_embed = (const float*)d_in[p++];
    Blk bb1[NBLOCKS], bb2[NBLOCKS];
    for (int i = 0; i < NBLOCKS; ++i) {
        bb1[i].fc1 = readCP(d_in, p); bb1[i].mr = readCP(d_in, p); bb1[i].fc2 = readCP(d_in, p);
        bb1[i].ffn1 = readCP(d_in, p); bb1[i].ffn2 = readCP(d_in, p);
    }
    for (int i = 0; i < NBLOCKS; ++i) {
        bb2[i].fc1 = readCP(d_in, p); bb2[i].mr = readCP(d_in, p); bb2[i].fc2 = readCP(d_in, p);
        bb2[i].ffn1 = readCP(d_in, p); bb2[i].ffn2 = readCP(d_in, p);
    }
    CP pred1 = readCP(d_in, p);
    CP pred2 = readCP(d_in, p);

    // -------- workspace bump allocator --------
    char* wptr = (char*)d_ws;
    auto alloc = [&](size_t bytes) -> void* {
        void* q = (void*)wptr;
        wptr += (bytes + 255) & ~(size_t)255;
        return q;
    };
    WS ws;
    const int Bb = BATCH, Cc = CCH, Nn = NSEG;
    ws.featA = (float*)alloc((size_t)Bb * 11 * Nn * 4);
    ws.x1    = (float*)alloc((size_t)Bb * Cc * Nn * 4);
    ws.x2    = (float*)alloc((size_t)Bb * Cc * Nn * 4);
    ws.y     = (float*)alloc((size_t)Bb * Cc * Nn * 4);
    ws.fn    = (float*)alloc((size_t)Bb * Cc * Nn * 4);
    ws.fnT   = (float*)alloc((size_t)Bb * Cc * Nn * 4);
    ws.sim   = (float*)alloc((size_t)Bb * Nn * Nn * 4);
    ws.idx   = (int*)  alloc((size_t)Bb * Nn * KMAX * 4);
    ws.big   = (float*)alloc((size_t)Bb * 4 * Cc * Nn * 4);  // z2+z3 / t4
    ws.tA    = (float*)alloc((size_t)2097152 * 4);
    ws.tB    = (float*)alloc((size_t)2097152 * 4);
    ws.xm    = (float*)alloc((size_t)Cc * Bb * 4);
    ws.p1    = (float*)alloc((size_t)1024 * Bb * 4);
    ws.p2    = (float*)alloc((size_t)NCLS * Bb * 4);
    ws.bnm   = (float*)alloc(1024 * 4);
    ws.bnr   = (float*)alloc(1024 * 4);

    // -------- branch 1: segment features + 1x1 stem --------
    features_kernel<<<Bb, 256, 0, stream>>>(inputs, segmaps, ws.featA);
    {
        int dims[6] = {11, 24, 48, 96, 192, 192};
        const float* cur = ws.featA;
        for (int li = 0; li < 5; ++li) {
            float* out = (li & 1) ? ws.x1 : ws.y;
            int O = dims[li + 1], I = dims[li];
            launch_gemm(slic[li].w, 0, cur, (ll)I * Nn, out, (ll)O * Nn, slic[li].b, O, Nn, I, Bb, stream);
            if (li < 4) {
                launch_bn(out, out, slic[li].g, slic[li].bb, nullptr, 1, Bb, O, Nn, ws.bnm, ws.bnr, stream);
                cur = out;
            } else {
                launch_bn(out, ws.x1, slic[li].g, slic[li].bb, nullptr, 0, Bb, O, Nn, ws.bnm, ws.bnr, stream);
            }
        }
    }

    // -------- branch 2: 3x3 conv stem --------
    {
        struct LC { int ci, h, w, co, ho, wo, st, pd; };
        LC cfg[5] = {
            {3,   64, 64, 24,  32, 32, 2, 1},
            {24,  32, 32, 48,  16, 16, 2, 1},
            {48,  16, 16, 96,  16, 16, 1, 1},
            {96,  16, 16, 192, 16, 16, 1, 1},
            {192, 16, 16, 192, 14, 14, 1, 0},
        };
        const float* cur = inputs;
        for (int li = 0; li < 5; ++li) {
            float* out = (li == 4) ? ws.x2 : ((li & 1) ? ws.tB : ws.tA);
            LC& c = cfg[li];
            ll total = (ll)Bb * c.co * c.ho * c.wo;
            conv3x3_kernel<<<(int)((total + 255) / 256), 256, 0, stream>>>(
                cur, tiny[li].w, tiny[li].b, out, c.ci, c.h, c.w, c.co, c.ho, c.wo, c.st, c.pd, total);
            launch_bn(out, out, tiny[li].g, tiny[li].bb, nullptr, (li < 4) ? 1 : 0,
                      Bb, c.co, c.ho * c.wo, ws.bnm, ws.bnr, stream);
            cur = out;
        }
        ll total = (ll)Bb * Cc * Nn;
        add_pos_kernel<<<(int)((total + 255) / 256), 256, 0, stream>>>(ws.x2, pos_embed, Cc * Nn, total);
    }

    // -------- 12 grapher + ffn blocks per branch --------
    static const int knn[NBLOCKS] = {9, 9, 10, 11, 12, 13, 13, 14, 15, 16, 17, 18};
    for (int i = 0; i < NBLOCKS; ++i) {
        run_block(ws.x1, bb1[i], knn[i], ws, stream);
        run_block(ws.x2, bb2[i], knn[i], ws, stream);
    }

    // -------- head --------
    head_mean_kernel<<<(Bb * Cc + 255) / 256, 256, 0, stream>>>(ws.x1, ws.x2, ws.xm, Cc, Nn, Bb);
    launch_gemm(pred1.w, 0, ws.xm, 0, ws.p1, 0, pred1.b, 1024, Bb, Cc, 1, stream);
    launch_bn(ws.p1, ws.p1, pred1.g, pred1.bb, nullptr, 1, 1, 1024, Bb, ws.bnm, ws.bnr, stream);
    launch_gemm(pred2.w, 0, ws.p1, 0, ws.p2, 0, pred2.b, NCLS, Bb, 1024, 1, stream);
    final_out_kernel<<<(Bb * NCLS + 255) / 256, 256, 0, stream>>>(ws.p2, (float*)d_out);
}
